// Router_76304388981193
// MI455X (gfx1250) — matrix operand order repaired and checked by reading, not compiled, not run on hardware
//
#include <hip/hip_runtime.h>
#include <hip/hip_bf16.h>
#include <stdint.h>

#define D_MODEL     4096
#define NUM_EXPERTS 64
#define TOKENS      (4 * 4096)          // B*S = 16384
#define TOK_PER_WG  64                  // 4 token-groups * 16 tokens
#define LDS_STRIDE  65                  // 64 + 1 pad: avoid bank conflicts
#define K_HALF      (D_MODEL / 2)       // 2 waves split the K reduction

typedef __attribute__((ext_vector_type(16))) __bf16          v16bf;
typedef __attribute__((ext_vector_type(8)))  unsigned        v8u;
typedef __attribute__((ext_vector_type(8)))  float           v8f;

// packed f32x2 -> bf16x2 via the native gfx1250 instruction (RNE).
static __device__ __forceinline__ unsigned pkbf(float a, float b) {
    unsigned r;
    asm("v_cvt_pk_bf16_f32 %0, %1, %2" : "=v"(r) : "v"(a), "v"(b));
    return r;
}

static __device__ __forceinline__ v8f wmma_bf16(v16bf a, v8u bu, v8f c) {
    return __builtin_amdgcn_wmma_f32_16x16x32_bf16(
        false, a, false, __builtin_bit_cast(v16bf, bu), (short)0, c, false, false);
}

// ---------------------------------------------------------------------------
// Kernel 1: convert W [64, 4096] fp32 -> bf16 into workspace (512 KB).
// Tiny (1 MB read); W then lives in the 192 MB L2 for the main kernel.
// ---------------------------------------------------------------------------
__global__ void __launch_bounds__(256)
convert_W_bf16(const float* __restrict__ W, unsigned* __restrict__ Wb) {
    int i = blockIdx.x * 256 + threadIdx.x;         // 65536 threads, 4 elems each
    const float4 v = reinterpret_cast<const float4*>(W)[i];
    uint2 o;
    o.x = pkbf(v.x, v.y);
    o.y = pkbf(v.z, v.w);
    reinterpret_cast<uint2*>(Wb)[i] = o;
}

// ---------------------------------------------------------------------------
// Kernel 2: router GEMM (bf16 WMMA, f32 accum) + bias + top-2 + softmax.
// Block = 256 threads (8 waves): 4 token-groups x 2 K-halves.
// Wave (tg, kh) computes tokens [tg*16, tg*16+16) over K in [kh*2048, +2048),
// partials summed in LDS. 2048 waves device-wide for latency hiding.
// ---------------------------------------------------------------------------
__global__ void __launch_bounds__(256)
router_kernel(const float* __restrict__ x,
              const unsigned short* __restrict__ Wb,
              const float* __restrict__ bias,
              float* __restrict__ outW,
              float* __restrict__ outI) {
    __shared__ float slog[TOK_PER_WG * LDS_STRIDE];

    const int tid  = threadIdx.x;
    const int wave = tid >> 5;
    const int tg   = wave >> 1;       // token group 0..3
    const int kh   = wave & 1;        // K half 0..1
    const int lane = tid & 31;
    const int half = lane >> 4;       // which 16-lane half of the wave
    const int col  = lane & 15;       // M-row (A/C) or N-col (B/C) within tile

    // zero the logits accumulation buffer
    for (int i = tid; i < TOK_PER_WG * LDS_STRIDE; i += 256) slog[i] = 0.0f;
    __syncthreads();

    const int tokBase = blockIdx.x * TOK_PER_WG + tg * 16;
    const float* xrow = x + (size_t)(tokBase + col) * D_MODEL;
    const unsigned short* wcol = Wb + (size_t)col * D_MODEL + 16 * half;

    // Accumulators: 4 N-tiles of 16 experts; K-half 0 folds the bias into C.
    // C layout: elem r, lanes0-15 -> (M=r, N=lane); lanes16-31 -> (M=r+8, N=lane-16)
    v8f acc[4];
#pragma unroll
    for (int nt = 0; nt < 4; ++nt) {
        float bv = kh ? 0.0f : bias[nt * 16 + col];
#pragma unroll
        for (int r = 0; r < 8; ++r) acc[nt][r] = bv;
    }

    const int kbeg = kh * K_HALF;
    const int kend = kbeg + K_HALF;
#pragma unroll 2
    for (int kb = kbeg; kb < kend; kb += 32) {
        // keep the x stream warm ~2KB ahead (speculative, counter-free)
        __builtin_prefetch(xrow + kb + 512, 0, 3);

        // ---- A tile loads: x[row, kb+16h..+15] (contiguous) ----
        const float4* pa = reinterpret_cast<const float4*>(xrow + kb + 16 * half);
        const float4 a0 = pa[0];
        const float4 a1 = pa[1];
        const float4* pc = reinterpret_cast<const float4*>(xrow + kb + 16 * half + 8);
        const float4 a2 = pc[0];
        const float4 a3 = pc[1];

        // ---- 4 independent B tiles: keep all 8 loads in flight ----
        const unsigned short* wp = wcol + kb;
        const v8u bu0 = *reinterpret_cast<const v8u*>(wp);
        const v8u bu1 = *reinterpret_cast<const v8u*>(wp + 16 * D_MODEL);
        const v8u bu2 = *reinterpret_cast<const v8u*>(wp + 32 * D_MODEL);
        const v8u bu3 = *reinterpret_cast<const v8u*>(wp + 48 * D_MODEL);

        v8u au;
        au[0] = pkbf(a0.x, a0.y); au[1] = pkbf(a0.z, a0.w);
        au[2] = pkbf(a1.x, a1.y); au[3] = pkbf(a1.z, a1.w);
        au[4] = pkbf(a2.x, a2.y); au[5] = pkbf(a2.z, a2.w);
        au[6] = pkbf(a3.x, a3.y); au[7] = pkbf(a3.z, a3.w);
        v16bf amat = __builtin_bit_cast(v16bf, au);

        acc[0] = wmma_bf16(amat, bu0, acc[0]);
        acc[1] = wmma_bf16(amat, bu1, acc[1]);
        acc[2] = wmma_bf16(amat, bu2, acc[2]);
        acc[3] = wmma_bf16(amat, bu3, acc[3]);
    }
    __syncthreads();   // zero-init visible; both K-halves done writing VGPRs

    // ---- combine partials in LDS (ds_add_f32): each (token, expert) gets
    //      one add from each K-half wave ----
#pragma unroll
    for (int nt = 0; nt < 4; ++nt) {
#pragma unroll
        for (int r = 0; r < 8; ++r) {
            int m = tg * 16 + r + 8 * half;       // token within WG
            atomicAdd(&slog[m * LDS_STRIDE + nt * 16 + col], acc[nt][r]);
        }
    }
    __syncthreads();

    // ---- top-2 + softmax-of-2, one thread per token (branchless scan) ----
    if (tid < TOK_PER_WG) {
        const float* row = &slog[tid * LDS_STRIDE];
        float b1 = -3.402823466e38f, b2 = -3.402823466e38f;
        int   i1 = 0, i2 = 0;
#pragma unroll 8
        for (int e = 0; e < NUM_EXPERTS; ++e) {
            float v  = row[e];
            bool  g1 = v > b1;                 // strict > keeps lowest index on ties
            bool  g2 = v > b2;
            float nb2 = g1 ? b1 : (g2 ? v : b2);
            int   ni2 = g1 ? i1 : (g2 ? e : i2);
            b1 = g1 ? v : b1;
            i1 = g1 ? e : i1;
            b2 = nb2;
            i2 = ni2;
        }
        float w1 = 1.0f / (1.0f + __expf(b2 - b1));  // softmax over {b1,b2}
        float w2 = 1.0f - w1;

        int t = blockIdx.x * TOK_PER_WG + tid;
        outW[t * 2 + 0] = w1;
        outW[t * 2 + 1] = w2;
        outI[t * 2 + 0] = (float)i1;   // top_k indices, promoted to f32 on concat
        outI[t * 2 + 1] = (float)i2;
    }
}

// ---------------------------------------------------------------------------
extern "C" void kernel_launch(void* const* d_in, const int* in_sizes, int n_in,
                              void* d_out, int out_size, void* d_ws, size_t ws_size,
                              hipStream_t stream) {
    const float* x    = (const float*)d_in[0];   // [4, 4096, 4096]
    const float* W    = (const float*)d_in[1];   // [64, 4096]
    const float* bias = (const float*)d_in[2];   // [64]

    unsigned* Wb = (unsigned*)d_ws;              // 512 KB bf16 W (pair-packed)

    float* outW = (float*)d_out;                 // [16384, 2] weights
    float* outI = outW + (size_t)TOKENS * 2;     // [16384, 2] indices (as f32)

    // Stage 1: W fp32 -> bf16 (65536 threads, 4 elems each)
    convert_W_bf16<<<256, 256, 0, stream>>>(W, Wb);

    // Stage 2: fused GEMM + top-2 + softmax. 256 WGs x 256 threads (8 waves).
    router_kernel<<<TOKENS / TOK_PER_WG, 256, 0, stream>>>(
        x, (const unsigned short*)Wb, bias, outW, outI);
}
